// NextFrameClassifier_8796093022493
// MI455X (gfx1250) — compile-verified, matrix operand order checked by reading, not run on hardware
//
#include <hip/hip_runtime.h>
#include <hip/hip_bf16.h>
#include <stdint.h>

typedef __attribute__((ext_vector_type(16))) _Float16 v16h;
typedef __attribute__((ext_vector_type(8)))  _Float16 v8h;
typedef __attribute__((ext_vector_type(8)))  float    v8f;
typedef __attribute__((ext_vector_type(4)))  unsigned v4u;
typedef __attribute__((ext_vector_type(8)))  unsigned v8u;

constexpr int Bc = 8;
constexpr int Tc = 2048;
constexpr int Mc = 2048;   // static segment upper bound
constexpr int MELc = 80;
constexpr int Dc = 64;
constexpr float EPSc = 1e-6f;
constexpr float THRESc = 0.04f;
constexpr float SCALEc = 100000.0f;

// LDS byte offset of a generic pointer that points into __shared__ memory.
__device__ inline uint32_t lds_addr(const void* p) {
    return (uint32_t)(uintptr_t)(__attribute__((address_space(3))) const void*)p;
}

// ---------------------------------------------------------------------------
// f32 -> f16 pack with K padding (weights, mel)
// ---------------------------------------------------------------------------
__global__ void pack_pad_f16(const float* __restrict__ src, _Float16* __restrict__ dst,
                             int rows, int Ks, int Kd) {
    int idx = blockIdx.x * 256 + threadIdx.x;
    if (idx >= rows * Kd) return;
    int r = idx / Kd, k = idx % Kd;
    dst[idx] = (_Float16)((k < Ks) ? src[r * Ks + k] : 0.0f);
}

// ---------------------------------------------------------------------------
// im2col for k=3, pad=1 conv: src [B][T][C] f16 -> dst [B*T][Kd] f16
// ---------------------------------------------------------------------------
__global__ void im2col3(const _Float16* __restrict__ src, _Float16* __restrict__ dst,
                        int C, int Kd) {
    int idx = blockIdx.x * 256 + threadIdx.x;
    int total = Bc * Tc * Kd;
    if (idx >= total) return;
    int row = idx / Kd, k = idx % Kd;
    int b = row / Tc, t = row % Tc;
    _Float16 v = (_Float16)0.0f;
    if (k < 3 * C) {
        int ci = k / 3, kk = k % 3;
        int tt = t + kk - 1;
        if (tt >= 0 && tt < Tc) v = src[((size_t)(b * Tc + tt)) * C + ci];
    }
    dst[idx] = v;
}

// ---------------------------------------------------------------------------
// Tiled conv GEMM with CDNA5 data movers.
// out[n][m] = act( sum_k W[m][k]*X[n][k] + bias[m] ),  Mout fixed = 256.
// Block: 256 threads (8 waves). Output tile: 256(M) x 64(N).
// Per K-chunk of 32:
//   - wave 0 issues a TDM tensor_load_to_lds of the W panel [256 x 32] (2D D#)
//   - every thread issues one global_load_async_to_lds_b128 of the X panel [64 x 32]
//   - each wave computes a 32x64 register tile = 8 WMMAs from LDS fragments
// ---------------------------------------------------------------------------
__global__ void wmma_gemm_conv(const _Float16* __restrict__ W, const _Float16* __restrict__ X,
                               const float* __restrict__ bias, _Float16* __restrict__ outH,
                               int K, int relu) {
    __shared__ _Float16 ldsW[256 * 32];   // 16 KB weight panel
    __shared__ _Float16 ldsX[64 * 32];    // 4 KB activation panel

    const int tid   = threadIdx.x;
    const int lane  = tid & 31;
    const int half  = lane >> 4;
    const int idx15 = lane & 15;
    const int m0    = (tid >> 5) * 32;            // 32 M-rows per wave
    const int n_base = blockIdx.x * 64;

    const uint32_t ldsW_base = lds_addr(ldsW);
    const uint32_t ldsX_base = lds_addr(ldsX);
    const int wave0 = (__builtin_amdgcn_readfirstlane(tid) < 32);  // scalar branch

    // async source for this thread: one b128 of the X panel per chunk
    const int xrow = tid >> 2, xseg = tid & 3;
    const _Float16* xsrc = X + (size_t)(n_base + xrow) * K + xseg * 8;
    const uint32_t  xdst = ldsX_base + (uint32_t)(xrow * 32 + xseg * 8) * 2u;

    v8f acc[2][4] = {};

    for (int k0 = 0; k0 < K; k0 += 32) {
        if (wave0) {
            // ---- TDM descriptor: 2D tile 32(k) x 256(m) of W, element = 2B ----
            unsigned long long ga = (unsigned long long)(uintptr_t)W + (unsigned long long)k0 * 2ull;
            v4u g0;
            g0[0] = 1u;                                   // count=1 (valid user D#)
            g0[1] = ldsW_base;                            // lds_addr (bytes)
            g0[2] = (unsigned)(ga & 0xFFFFFFFFull);       // global_addr[31:0]
            g0[3] = (unsigned)((ga >> 32) & 0x01FFFFFFull) | (2u << 30);  // addr[56:32], type=2
            v8u g1;
            g1[0] = 1u << 16;                             // data_size = 2 bytes
            g1[1] = ((unsigned)K & 0xFFFFu) << 16;        // tensor_dim0[15:0]
            g1[2] = (((unsigned)K >> 16) & 0xFFFFu) | (256u << 16);  // dim0 hi | tensor_dim1=256
            g1[3] = 32u << 16;                            // tile_dim0 = 32
            g1[4] = 256u;                                 // tile_dim1 = 256 (tile_dim2 = 0)
            g1[5] = (unsigned)K;                          // tensor_dim0_stride[31:0]
            g1[6] = 0u;
            g1[7] = 0u;
            asm volatile("tensor_load_to_lds %0, %1" :: "s"(g0), "s"(g1) : "memory");
            __builtin_amdgcn_s_wait_tensorcnt(0);
        }
        // ---- async copy of X panel: 256 threads x 16B = 64 rows x 64B ----
        asm volatile("global_load_async_to_lds_b128 %0, %1, off"
                     :: "v"(xdst), "v"(xsrc + k0) : "memory");
        asm volatile("s_wait_asynccnt 0x0" ::: "memory");
        __syncthreads();

        union { v16h v; v8h h[2]; } a[2], b[4];
#pragma unroll
        for (int i = 0; i < 2; ++i) {
            const _Float16* ap = ldsW + (m0 + i * 16 + idx15) * 32 + half * 8;
            a[i].h[0] = *(const v8h*)ap;
            a[i].h[1] = *(const v8h*)(ap + 16);
        }
#pragma unroll
        for (int j = 0; j < 4; ++j) {
            const _Float16* bp = ldsX + (j * 16 + idx15) * 32 + half * 8;
            b[j].h[0] = *(const v8h*)bp;
            b[j].h[1] = *(const v8h*)(bp + 16);
        }
#pragma unroll
        for (int i = 0; i < 2; ++i)
#pragma unroll
            for (int j = 0; j < 4; ++j)
                acc[i][j] = __builtin_amdgcn_wmma_f32_16x16x32_f16(
                    false, a[i].v, false, b[j].v, (short)0, acc[i][j], false, false);
        __syncthreads();
    }

#pragma unroll
    for (int i = 0; i < 2; ++i)
#pragma unroll
        for (int j = 0; j < 4; ++j) {
            int n = n_base + j * 16 + idx15;
#pragma unroll
            for (int r = 0; r < 8; ++r) {
                int m = m0 + i * 16 + r + half * 8;
                float v = acc[i][j][r] + bias[m];
                if (relu) v = fmaxf(v, 0.0f);
                outH[(size_t)n * 256 + m] = (_Float16)v;
            }
        }
}

// ---------------------------------------------------------------------------
// Generic one-wave-per-tile WMMA GEMM (small layers): K%32==0, Mout%16==0.
// ---------------------------------------------------------------------------
__global__ void wmma_gemm(const _Float16* __restrict__ W, const _Float16* __restrict__ X,
                          const float* __restrict__ bias,
                          float* __restrict__ outF, _Float16* __restrict__ outH,
                          int N, int Mout, int K, int relu) {
    int lane  = threadIdx.x & 31;
    int half  = lane >> 4;
    int idx15 = lane & 15;
    int n0 = blockIdx.x * 16;
    int m0 = blockIdx.y * 16;

    const _Float16* wrow = W + (size_t)(m0 + idx15) * K + half * 8;
    const _Float16* xrow = X + (size_t)(n0 + idx15) * K + half * 8;

    v8f acc = {};
    for (int k0 = 0; k0 < K; k0 += 32) {
        union { v16h v; v8h h[2]; } a, b;
        a.h[0] = *(const v8h*)(wrow + k0);
        a.h[1] = *(const v8h*)(wrow + k0 + 16);
        b.h[0] = *(const v8h*)(xrow + k0);
        b.h[1] = *(const v8h*)(xrow + k0 + 16);
        if (k0 + 32 < K) __builtin_prefetch(xrow + k0 + 32, 0, 0);
        acc = __builtin_amdgcn_wmma_f32_16x16x32_f16(
            false, a.v, false, b.v, (short)0, acc, false, false);
    }

    int n = n0 + idx15;
#pragma unroll
    for (int r = 0; r < 8; ++r) {
        int m = m0 + r + half * 8;
        float v = acc[r] + (bias ? bias[m] : 0.0f);
        if (relu) v = fmaxf(v, 0.0f);
        if (outF) outF[(size_t)n * Mout + m] = v;
        if (outH) outH[(size_t)n * Mout + m] = (_Float16)v;
    }
}

// ---------------------------------------------------------------------------
// Per-batch boundary detector + block scan. One block (256 threads) per b.
// ---------------------------------------------------------------------------
__global__ void boundary_kernel(const float* __restrict__ latent,
                                const int* __restrict__ lengths,
                                float* __restrict__ pos_pred, float* __restrict__ neg_pred,
                                float* __restrict__ d_out, float* __restrict__ pred_out,
                                float* __restrict__ mask_out,
                                float* __restrict__ seg_acc, int* __restrict__ sid_out) {
    __shared__ float s_sim[Tc];
    __shared__ float s_d[Tc];
    __shared__ float s_red[256];
    __shared__ float s_sum[256];

    int b = blockIdx.x;
    int tid = threadIdx.x;
    int len = lengths[b];
    const float* L = latent + (size_t)b * Tc * Dc;

    float lmin = 3.4e38f, lmax = -3.4e38f;
    for (int t = tid; t < Tc - 1; t += 256) {
        const float* a = L + (size_t)t * Dc;
        const float* c = a + Dc;
        float dot = 0.f, na = 0.f, nb = 0.f;
#pragma unroll 8
        for (int k = 0; k < Dc; ++k) { float x = a[k], y = c[k]; dot += x * y; na += x * x; nb += y * y; }
        float sim = dot / fmaxf(sqrtf(na) * sqrtf(nb), EPSc);
        s_sim[t] = sim;
        pos_pred[b * (Tc - 1) + t] = sim;
        neg_pred[b * (Tc - 1) + t] = na / fmaxf(na, EPSc);
        if (t < len - 1) { lmin = fminf(lmin, sim); lmax = fmaxf(lmax, sim); }
    }
    s_red[tid] = lmin; __syncthreads();
    for (int s = 128; s > 0; s >>= 1) { if (tid < s) s_red[tid] = fminf(s_red[tid], s_red[tid + s]); __syncthreads(); }
    float smin = s_red[0]; __syncthreads();
    s_red[tid] = lmax; __syncthreads();
    for (int s = 128; s > 0; s >>= 1) { if (tid < s) s_red[tid] = fmaxf(s_red[tid], s_red[tid + s]); __syncthreads(); }
    float smax = s_red[0]; __syncthreads();

    float inv = 1.0f / (smax - smin + EPSc);
    for (int t = tid; t < Tc; t += 256) {
        float v = 0.0f;
        if (t >= 1 && t < len) v = 1.0f - (s_sim[t - 1] - smin) * inv;
        s_d[t] = v;
    }
    __syncthreads();

    float pr[8];
    float local_sum = 0.0f;
    int base = tid * 8;
#pragma unroll
    for (int i = 0; i < 8; ++i) {
        int t = base + i;
        float d = s_d[t];
        bool mk = (t < len);
        float left   = mk ? ((t >= 1) ? s_d[t - 1] : 0.0f) : 0.0f;
        float right  = (t == 0) ? 0.0f : ((t < Tc - 1) ? s_d[t + 1] : 0.0f);
        float left2  = mk ? ((t >= 2) ? s_d[t - 2] : 0.0f) : 0.0f;
        float right2 = (t == 0) ? 0.0f : ((t < Tc - 2) ? s_d[t + 2] : 0.0f);
        float p1 = fminf(fmaxf(d - left, 0.0f), fmaxf(d - right, 0.0f));
        float p2 = fminf(fmaxf(d - left2, 0.0f), fmaxf(d - right2, 0.0f));
        float p  = fminf(fmaxf(fmaxf(p1, p2) - THRESc, 0.0f), p1);
        float pb = tanhf(1e9f * p);
        pr[i] = pb;
        local_sum += pb;
        d_out[b * Tc + t]    = d;
        pred_out[b * Tc + t] = pb;
        mask_out[b * Tc + t] = mk ? 1.0f : 0.0f;
    }
    s_sum[tid] = local_sum;
    __syncthreads();
    if (tid == 0) {
        float run = 0.0f;
        for (int i = 0; i < 256; ++i) { float v = s_sum[i]; s_sum[i] = run; run += v; }
    }
    __syncthreads();
    float run = s_sum[tid];
#pragma unroll
    for (int i = 0; i < 8; ++i) {
        int t = base + i;
        run += pr[i];
        bool mk = (t < len);
        seg_acc[b * Tc + t] = mk ? run : __builtin_inff();
        int s = -1;
        if (mk) {
            s = __float2int_rn(run);
            s = (s < 0) ? 0 : ((s > Mc - 1) ? Mc - 1 : s);
        }
        sid_out[b * Tc + t] = s;
    }
}

// ---------------------------------------------------------------------------
// Segment scatter-mean (W_int is a one-hot in m == seg_acc[t])
// ---------------------------------------------------------------------------
__global__ void seg_scatter(const float* __restrict__ latent, const int* __restrict__ sid,
                            float* __restrict__ segsum, int* __restrict__ segcnt) {
    int frame = blockIdx.x;
    int dd = threadIdx.x;
    int s = sid[frame];
    if (s < 0) return;
    int b = frame / Tc;
    atomicAdd(&segsum[((size_t)(b * Mc + s)) * Dc + dd], latent[(size_t)frame * Dc + dd]);
    if (dd == 0) atomicAdd(&segcnt[b * Mc + s], 1);
}

__global__ void seg_mean(const float* __restrict__ segsum, const int* __restrict__ segcnt,
                         _Float16* __restrict__ meanH, int* __restrict__ durations) {
    int idx = blockIdx.x * 256 + threadIdx.x;
    if (idx >= Bc * Mc * Dc) return;
    int row = idx / Dc;
    int cnt = segcnt[row];
    meanH[idx] = (_Float16)(segsum[idx] / ((float)cnt + EPSc));
    if ((idx % Dc) == 0) durations[row] = cnt;
}

__global__ void upsample_k(const float* __restrict__ segrep, const int* __restrict__ sid,
                           float* __restrict__ frameF, _Float16* __restrict__ frameH) {
    int frame = blockIdx.x;
    int dd = threadIdx.x;
    int b = frame / Tc;
    int s = sid[frame];
    if (s < 0) s = 0;
    float v = segrep[((size_t)(b * Mc + s)) * Dc + dd];
    frameF[(size_t)frame * Dc + dd] = v;
    frameH[(size_t)frame * Dc + dd] = (_Float16)v;
}

// ---------------------------------------------------------------------------
// Streaming V / W_int writer (bandwidth-dominant 268MB of output)
// ---------------------------------------------------------------------------
__global__ void vw_kernel(const float* __restrict__ segacc,
                          float* __restrict__ Vout, float* __restrict__ Wout) {
    long long idx = (long long)blockIdx.x * 256 + threadIdx.x;
    int t = (int)(idx % Tc);
    long long r = idx / Tc;
    int m = (int)(r % Mc);
    int b = (int)(r / Mc);
    float a = segacc[b * Tc + t];
    float v = (float)m - a;
    float w = 1.0f - tanhf(SCALEc * fabsf(v));
    Vout[idx] = v;
    Wout[idx] = w;
}

// ---------------------------------------------------------------------------
// Host-side orchestration
// ---------------------------------------------------------------------------
extern "C" void kernel_launch(void* const* d_in, const int* in_sizes, int n_in,
                              void* d_out, int out_size, void* d_ws, size_t ws_size,
                              hipStream_t stream) {
    const float* mel     = (const float*)d_in[0];
    const int*   lengths = (const int*)  d_in[1];
    const float* c1w = (const float*)d_in[2];  const float* c1b = (const float*)d_in[3];
    const float* c2w = (const float*)d_in[4];  const float* c2b = (const float*)d_in[5];
    const float* pjw = (const float*)d_in[6];  const float* pjb = (const float*)d_in[7];
    const float* s1w = (const float*)d_in[8];  const float* s1b = (const float*)d_in[9];
    const float* s2w = (const float*)d_in[10]; const float* s2b = (const float*)d_in[11];
    const float* a1w = (const float*)d_in[12]; const float* a1b = (const float*)d_in[13];
    const float* a2w = (const float*)d_in[14]; const float* a2b = (const float*)d_in[15];

    float* out = (float*)d_out;
    const long long O_recon = 0;
    const long long O_frame = 1310720;
    const long long O_seg   = 2359296;
    const long long O_dur   = 3407872;
    const long long O_lat   = 3424256;
    const long long O_pred  = 4472832;
    const long long O_mask  = 4489216;
    const long long O_d     = 4505600;
    const long long O_V     = 4521984;
    const long long O_W     = 38076416;
    const long long O_pos   = 71630848;
    const long long O_neg   = 71647224;

    char* w = (char*)d_ws;
    _Float16* melh  = (_Float16*)w; w += (size_t)Bc*Tc*MELc*2;
    _Float16* W1h   = (_Float16*)w; w += 256*256*2;
    _Float16* W2h   = (_Float16*)w; w += 256*768*2;
    _Float16* PWh   = (_Float16*)w; w += 64*256*2;
    _Float16* S1h   = (_Float16*)w; w += 128*64*2;
    _Float16* S2h   = (_Float16*)w; w += 64*128*2;
    _Float16* A1h   = (_Float16*)w; w += 32*64*2;
    _Float16* A2h   = (_Float16*)w; w += 80*32*2;
    _Float16* X1im  = (_Float16*)w; w += (size_t)Bc*Tc*256*2;
    _Float16* Y1    = (_Float16*)w; w += (size_t)Bc*Tc*256*2;
    _Float16* X2im  = (_Float16*)w; w += (size_t)Bc*Tc*768*2;
    _Float16* Y2    = (_Float16*)w; w += (size_t)Bc*Tc*256*2;
    float*    segacc= (float*)w;    w += (size_t)Bc*Tc*4;
    int*      sid   = (int*)w;      w += (size_t)Bc*Tc*4;
    float*    segsum= (float*)w;    w += (size_t)Bc*Mc*Dc*4;
    int*      segcnt= (int*)w;      w += (size_t)Bc*Mc*4;
    _Float16* meanH = (_Float16*)w; w += (size_t)Bc*Mc*Dc*2;
    _Float16* segH  = (_Float16*)w; w += (size_t)Bc*Mc*128*2;
    _Float16* frameH= (_Float16*)w; w += (size_t)Bc*Tc*Dc*2;
    _Float16* aeH   = (_Float16*)w; w += (size_t)Bc*Tc*32*2;

    auto blocks = [](long long n) { return (unsigned)((n + 255) / 256); };

    pack_pad_f16<<<blocks((long long)Bc*Tc*MELc), 256, 0, stream>>>(mel, melh, Bc*Tc, MELc, MELc);
    pack_pad_f16<<<blocks(256*256), 256, 0, stream>>>(c1w, W1h, 256, 240, 256);
    pack_pad_f16<<<blocks(256*768), 256, 0, stream>>>(c2w, W2h, 256, 768, 768);
    pack_pad_f16<<<blocks(64*256),  256, 0, stream>>>(pjw, PWh, 64, 256, 256);
    pack_pad_f16<<<blocks(128*64),  256, 0, stream>>>(s1w, S1h, 128, 64, 64);
    pack_pad_f16<<<blocks(64*128),  256, 0, stream>>>(s2w, S2h, 64, 128, 128);
    pack_pad_f16<<<blocks(32*64),   256, 0, stream>>>(a1w, A1h, 32, 64, 64);
    pack_pad_f16<<<blocks(80*32),   256, 0, stream>>>(a2w, A2h, 80, 32, 32);

    const int N = Bc * Tc;  // 16384

    // conv1 / conv2: TDM + async-LDS tiled WMMA GEMMs (Mout = 256)
    im2col3<<<blocks((long long)N*256), 256, 0, stream>>>(melh, X1im, MELc, 256);
    wmma_gemm_conv<<<N/64, 256, 0, stream>>>(W1h, X1im, c1b, Y1, 256, 1);
    im2col3<<<blocks((long long)N*768), 256, 0, stream>>>(Y1, X2im, 256, 768);
    wmma_gemm_conv<<<N/64, 256, 0, stream>>>(W2h, X2im, c2b, Y2, 768, 1);

    // projection -> latent f32
    wmma_gemm<<<dim3(N/16, 64/16), 32, 0, stream>>>(PWh, Y2, pjb, out + O_lat, nullptr, N, 64, 256, 0);

    boundary_kernel<<<Bc, 256, 0, stream>>>(out + O_lat, lengths,
                                            out + O_pos, out + O_neg,
                                            out + O_d, out + O_pred, out + O_mask,
                                            segacc, sid);

    hipMemsetAsync(segsum, 0, (size_t)Bc*Mc*Dc*4 + (size_t)Bc*Mc*4, stream);
    seg_scatter<<<N, 64, 0, stream>>>(out + O_lat, sid, segsum, segcnt);
    seg_mean<<<blocks((long long)Bc*Mc*Dc), 256, 0, stream>>>(segsum, segcnt, meanH,
                                                              (int*)(out + O_dur));

    wmma_gemm<<<dim3(N/16, 128/16), 32, 0, stream>>>(S1h, meanH, s1b, nullptr, segH, N, 128, 64, 1);
    wmma_gemm<<<dim3(N/16, 64/16),  32, 0, stream>>>(S2h, segH,  s2b, out + O_seg, nullptr, N, 64, 128, 0);

    upsample_k<<<N, 64, 0, stream>>>(out + O_seg, sid, out + O_frame, frameH);

    wmma_gemm<<<dim3(N/16, 32/16), 32, 0, stream>>>(A1h, frameH, a1b, nullptr, aeH, N, 32, 64, 1);
    wmma_gemm<<<dim3(N/16, 80/16), 32, 0, stream>>>(A2h, aeH, a2b, out + O_recon, nullptr, N, 80, 32, 0);

    vw_kernel<<<(unsigned)(((long long)Bc*Mc*Tc) / 256), 256, 0, stream>>>(segacc,
                                                                           out + O_V, out + O_W);
}
